// FullAttention_17789754540074
// MI455X (gfx1250) — matrix operand order repaired and checked by reading, not compiled, not run on hardware
//
#include <hip/hip_runtime.h>
#include <hip/hip_bf16.h>

// ---------------------------------------------------------------------------
// Sizes (fixed by the reference)
// ---------------------------------------------------------------------------
#define HID   768
#define HEADS 12
#define HDIM  64
#define MLP   3072
#define NFUSE (MLP + 3 * HID)   // 5376
#define GB    2                 // batch
#define SP    12                // spatial edge
#define SEQ   (SP * SP * SP)    // 1728
#define TOK   (GB * SEQ)        // 3456
#define KCAT  (HID + MLP / 2)   // 2304 (att(768) ++ ff_act(1536))

typedef _Float16 h8  __attribute__((ext_vector_type(8)));
typedef _Float16 h16 __attribute__((ext_vector_type(16)));
typedef float    f8  __attribute__((ext_vector_type(8)));

// ---------------------------------------------------------------------------
// WMMA fragment loads (wave32; this code's contiguous k order, same in A and B), branch-free.
// A (16x32 f16): lane m = L&15 ; lanes 0-15 hold K 0..15, lanes 16-31
// hold K 16..31  -> two 16B contiguous loads per lane.
// B (32x16 f16): lane n = L&15 ; lanes 0-15 hold K 0..15, lanes 16-31 hold
// K 16..31 -> one 32B contiguous load per lane.
// ---------------------------------------------------------------------------
__device__ __forceinline__ h16 ld_a(const _Float16* __restrict__ p) {
  h8 lo = *(const h8*)(p);
  h8 hi = *(const h8*)(p + 8);
  return __builtin_shufflevector(lo, hi, 0, 1, 2, 3, 4, 5, 6, 7,
                                          8, 9, 10, 11, 12, 13, 14, 15);
}
__device__ __forceinline__ h16 ld_b(const _Float16* __restrict__ p) {
  return *(const h16*)(p);
}

#define WMMA_F16(a, b, c) \
  __builtin_amdgcn_wmma_f32_16x16x32_f16(false, (a), false, (b), (short)0, (c), false, false)

// ---------------------------------------------------------------------------
// Generic WMMA GEMM:  C[M,N] = A[M,K] * B[N,K]^T (+bias[n])
// Wave tile 32x64 (2 A-frags x 4 B-frags, 8 f32 accumulators).
// Edge handling: OOB lanes CLAMP their row/col index (garbage A row m only
// pollutes D row m; garbage B col n only pollutes D col n; both are masked at
// the store) -> inner loop is branch-free, pure load+wmma, software-pipelined
// one k-step ahead. K must be a multiple of 32 (true for 768/64/1728/2304).
// Batched over blockIdx.z; C offset per z = (z/zdiv)*sC1 + (z%zdiv)*sC2.
// ---------------------------------------------------------------------------
template <int MW, int NW, bool OUT_HALF>
__global__ __launch_bounds__(MW * NW * 32) void wmma_gemm(
    const _Float16* __restrict__ A, long lda, long sAz,
    const _Float16* __restrict__ B, long ldb, long sBz,
    void* __restrict__ Cv, long ldc, int zdiv, long sC1, long sC2,
    const float* __restrict__ bias, int M, int N, int K) {
  const int z = blockIdx.z;
  A += (long)z * sAz;
  B += (long)z * sBz;
  const long coff = (long)(z / zdiv) * sC1 + (long)(z % zdiv) * sC2;

  const int lane = threadIdx.x & 31;
  const int wid  = threadIdx.x >> 5;
  const int wm   = wid % MW;
  const int wn   = wid / MW;
  const int m_base = blockIdx.y * (MW * 32) + wm * 32;
  const int n_base = blockIdx.x * (NW * 64) + wn * 64;
  const int l15    = lane & 15;
  const int a_koff = (lane >> 4) << 4;   // 0 or 16
  const int b_koff = (lane >> 4) << 4;   // 0 or 16

  // Clamped per-lane base pointers (hoisted out of the k-loop).
  const _Float16* pa0 = A + (long)min(m_base + l15,      M - 1) * lda + a_koff;
  const _Float16* pa1 = A + (long)min(m_base + 16 + l15, M - 1) * lda + a_koff;
  const _Float16* pb0 = B + (long)min(n_base + l15,      N - 1) * ldb + b_koff;
  const _Float16* pb1 = B + (long)min(n_base + 16 + l15, N - 1) * ldb + b_koff;
  const _Float16* pb2 = B + (long)min(n_base + 32 + l15, N - 1) * ldb + b_koff;
  const _Float16* pb3 = B + (long)min(n_base + 48 + l15, N - 1) * ldb + b_koff;

  f8 acc[2][4];
#pragma unroll
  for (int i = 0; i < 2; ++i)
#pragma unroll
    for (int j = 0; j < 4; ++j) acc[i][j] = {};

  // Prologue: fragments for k = 0.
  h16 a0 = ld_a(pa0), a1 = ld_a(pa1);
  h16 b0 = ld_b(pb0), b1 = ld_b(pb1), b2 = ld_b(pb2), b3 = ld_b(pb3);

  // Pipelined main loop: issue loads for k, compute WMMAs for k-32.
  for (int k = 32; k < K; k += 32) {
    h16 na0 = ld_a(pa0 + k), na1 = ld_a(pa1 + k);
    h16 nb0 = ld_b(pb0 + k), nb1 = ld_b(pb1 + k);
    h16 nb2 = ld_b(pb2 + k), nb3 = ld_b(pb3 + k);

    acc[0][0] = WMMA_F16(a0, b0, acc[0][0]);
    acc[0][1] = WMMA_F16(a0, b1, acc[0][1]);
    acc[0][2] = WMMA_F16(a0, b2, acc[0][2]);
    acc[0][3] = WMMA_F16(a0, b3, acc[0][3]);
    acc[1][0] = WMMA_F16(a1, b0, acc[1][0]);
    acc[1][1] = WMMA_F16(a1, b1, acc[1][1]);
    acc[1][2] = WMMA_F16(a1, b2, acc[1][2]);
    acc[1][3] = WMMA_F16(a1, b3, acc[1][3]);

    a0 = na0; a1 = na1; b0 = nb0; b1 = nb1; b2 = nb2; b3 = nb3;
  }

  // Epilogue k-step.
  acc[0][0] = WMMA_F16(a0, b0, acc[0][0]);
  acc[0][1] = WMMA_F16(a0, b1, acc[0][1]);
  acc[0][2] = WMMA_F16(a0, b2, acc[0][2]);
  acc[0][3] = WMMA_F16(a0, b3, acc[0][3]);
  acc[1][0] = WMMA_F16(a1, b0, acc[1][0]);
  acc[1][1] = WMMA_F16(a1, b1, acc[1][1]);
  acc[1][2] = WMMA_F16(a1, b2, acc[1][2]);
  acc[1][3] = WMMA_F16(a1, b3, acc[1][3]);

  // C/D layout: VGPR v holds row m = base + v + 8*(lane>=16), col = l15.
#pragma unroll
  for (int i = 0; i < 2; ++i) {
    const int mrow = m_base + i * 16 + ((lane >> 4) << 3);
#pragma unroll
    for (int j = 0; j < 4; ++j) {
      const int col = n_base + j * 16 + l15;
      if (col < N) {
        const float bv = bias ? bias[col] : 0.f;
#pragma unroll
        for (int v = 0; v < 8; ++v) {
          const int r = mrow + v;
          if (r < M) {
            const float val = acc[i][j][v] + bv;
            if (OUT_HALF)
              ((_Float16*)Cv)[coff + (long)r * ldc + col] = (_Float16)val;
            else
              ((float*)Cv)[coff + (long)r * ldc + col] = val;
          }
        }
      }
    }
  }
}

// ---------------------------------------------------------------------------
// f32 -> f16 weight conversion (optionally scatters into a wider row: used to
// concatenate w_attn ++ w_ff into one [768, 2304] weight for the fused
// attn-proj + ff-proj GEMM).
// ---------------------------------------------------------------------------
__global__ void cvt_kernel(const float* __restrict__ src, _Float16* __restrict__ dst,
                           long rows, int cols, int dstride, int doff) {
  const long i = (long)blockIdx.x * blockDim.x + threadIdx.x;
  if (i >= rows * (long)cols) return;
  const long r = i / cols;
  const int  c = (int)(i % cols);
  dst[r * dstride + doff + c] = (_Float16)src[i];
}

// ---------------------------------------------------------------------------
// RMS over spatial per (b,c), scale by norm1_w, transpose to token-major f16.
// ---------------------------------------------------------------------------
__global__ __launch_bounds__(256) void rmsnorm_kernel(
    const float* __restrict__ x, const float* __restrict__ w,
    _Float16* __restrict__ xn) {
  const int b = blockIdx.x / HID;
  const int c = blockIdx.x % HID;
  const float* xp = x + ((long)b * HID + c) * SEQ;
  float ss = 0.f;
  for (int s = threadIdx.x; s < SEQ; s += 256) {
    const float v = xp[s];
    ss += v * v;
  }
  for (int o = 16; o > 0; o >>= 1) ss += __shfl_xor(ss, o, 32);
  __shared__ float red[8];
  const int lane = threadIdx.x & 31, wid = threadIdx.x >> 5;
  if (lane == 0) red[wid] = ss;
  __syncthreads();
  if (wid == 0) {
    float t = (lane < 8) ? red[lane] : 0.f;
    for (int o = 4; o > 0; o >>= 1) t += __shfl_xor(t, o, 32);
    if (lane == 0) red[0] = t;
  }
  __syncthreads();
  const float scale = w[c] * rsqrtf(red[0] * (1.f / SEQ) + 1e-6f);
  for (int s = threadIdx.x; s < SEQ; s += 256)
    xn[((long)b * SEQ + s) * HID + c] = (_Float16)(xp[s] * scale);
}

// ---------------------------------------------------------------------------
// Per-(token, head): layernorm q/k over 64 dims, axial RoPE on dims [0,48),
// emit q16 (pre-scaled by 1/8), k16 [bh][s][d], vT16 [bh][d][s] (transposed
// so PV is an NT WMMA GEMM). One wave per (token,head); lane owns the
// rotation pair (2*lane, 2*lane+1) so rotate_half stays lane-local.
// ---------------------------------------------------------------------------
__global__ __launch_bounds__(256) void qkv_prep_kernel(
    const float* __restrict__ fused,
    const float* __restrict__ qg, const float* __restrict__ qb,
    const float* __restrict__ kg, const float* __restrict__ kb,
    _Float16* __restrict__ q16, _Float16* __restrict__ k16,
    _Float16* __restrict__ vT16) {
  const int gw   = blockIdx.x * 8 + (threadIdx.x >> 5);
  const int lane = threadIdx.x & 31;
  const int token = gw / HEADS;
  const int head  = gw % HEADS;
  const int b = token / SEQ;
  const int s = token % SEQ;
  const float* base = fused + (long)token * NFUSE;
  const int d0 = 2 * lane, d1 = d0 + 1;

  float q0 = base[MLP + head * HDIM + d0],           q1 = base[MLP + head * HDIM + d1];
  float k0 = base[MLP + HID + head * HDIM + d0],     k1 = base[MLP + HID + head * HDIM + d1];
  const float v0 = base[MLP + 2 * HID + head * HDIM + d0];
  const float v1 = base[MLP + 2 * HID + head * HDIM + d1];

  auto ln = [&](float& t0, float& t1, const float* g, const float* be) {
    float sum = t0 + t1;
    for (int o = 16; o > 0; o >>= 1) sum += __shfl_xor(sum, o, 32);
    const float mu = sum * (1.f / 64.f);
    float e0 = t0 - mu, e1 = t1 - mu;
    float var = e0 * e0 + e1 * e1;
    for (int o = 16; o > 0; o >>= 1) var += __shfl_xor(var, o, 32);
    const float inv = rsqrtf(var * (1.f / 64.f) + 1e-5f);
    t0 = e0 * inv * g[d0] + be[d0];
    t1 = e1 * inv * g[d1] + be[d1];
  };
  ln(q0, q1, qg, qb);
  ln(k0, k1, kg, kb);

  if (d0 < 48) {  // rot_dim = 3 * ROT_AXIS = 48
    const int axis = lane >> 3;   // 16 dims per axis -> 8 pairs per axis
    const int j    = lane & 7;    // freq index within axis
    const int pidx = (axis == 0) ? (s / 144) : (axis == 1) ? ((s / 12) % 12) : (s % 12);
    const float pos   = -1.f + 2.f * (float)pidx / 11.f;          // linspace(-1,1,12)
    const float basef = 3.14159265358979f * (1.f + 127.f * (float)j / 7.f);  // linspace(1,128,8)*pi
    float sf, cf;
    __sincosf(pos * basef, &sf, &cf);
    const float nq0 = q0 * cf - q1 * sf, nq1 = q1 * cf + q0 * sf;
    const float nk0 = k0 * cf - k1 * sf, nk1 = k1 * cf + k0 * sf;
    q0 = nq0; q1 = nq1; k0 = nk0; k1 = nk1;
  }

  const long bh = (long)b * HEADS + head;
  _Float16* qp = q16 + (bh * SEQ + s) * HDIM;
  _Float16* kp = k16 + (bh * SEQ + s) * HDIM;
  qp[d0] = (_Float16)(q0 * 0.125f);  // fold 1/sqrt(64) softmax scale into q
  qp[d1] = (_Float16)(q1 * 0.125f);
  kp[d0] = (_Float16)k0;
  kp[d1] = (_Float16)k1;
  _Float16* vp = vT16 + bh * (long)HDIM * SEQ;
  vp[(long)d0 * SEQ + s] = (_Float16)v0;
  vp[(long)d1 * SEQ + s] = (_Float16)v1;
}

// ---------------------------------------------------------------------------
// Row softmax over the f16 score matrix, in place; row staged in LDS (6.9 KB)
// so global traffic is one read + one write.
// ---------------------------------------------------------------------------
__global__ __launch_bounds__(256) void softmax_kernel(_Float16* __restrict__ p) {
  __shared__ float buf[SEQ];
  __shared__ float red[8];
  _Float16* row = p + (long)blockIdx.x * SEQ;
  const int tid = threadIdx.x, lane = tid & 31, wid = tid >> 5;

  float mx = -1e30f;
  for (int i = tid; i < SEQ; i += 256) {
    const float v = (float)row[i];
    buf[i] = v;
    mx = fmaxf(mx, v);
  }
  for (int o = 16; o > 0; o >>= 1) mx = fmaxf(mx, __shfl_xor(mx, o, 32));
  if (lane == 0) red[wid] = mx;
  __syncthreads();
  if (wid == 0) {
    float t = (lane < 8) ? red[lane] : -1e30f;
    for (int o = 4; o > 0; o >>= 1) t = fmaxf(t, __shfl_xor(t, o, 32));
    if (lane == 0) red[0] = t;
  }
  __syncthreads();
  mx = red[0];
  __syncthreads();

  float sum = 0.f;
  for (int i = tid; i < SEQ; i += 256) {
    const float e = __expf(buf[i] - mx);
    buf[i] = e;
    sum += e;
  }
  for (int o = 16; o > 0; o >>= 1) sum += __shfl_xor(sum, o, 32);
  if (lane == 0) red[wid] = sum;
  __syncthreads();
  if (wid == 0) {
    float t = (lane < 8) ? red[lane] : 0.f;
    for (int o = 4; o > 0; o >>= 1) t += __shfl_xor(t, o, 32);
    if (lane == 0) red[0] = t;
  }
  __syncthreads();
  const float inv = 1.f / red[0];
  for (int i = tid; i < SEQ; i += 256) row[i] = (_Float16)(buf[i] * inv);
}

// ---------------------------------------------------------------------------
// SwiGLU: actff[token][768 + j] = silu(gate) * ff_x  (fills the FF half of the
// concatenated activation buffer; the attention half is written by PV GEMM).
// ---------------------------------------------------------------------------
__global__ void silu_kernel(const float* __restrict__ fused,
                            _Float16* __restrict__ actff) {
  const long i = (long)blockIdx.x * blockDim.x + threadIdx.x;
  if (i >= (long)TOK * (MLP / 2)) return;
  const long token = i / (MLP / 2);
  const int  j     = (int)(i % (MLP / 2));
  const float* r = fused + token * NFUSE;
  const float g  = r[MLP / 2 + j];   // gate = ff[..., 1536:3072]
  const float xx = r[j];             // ff_x = ff[..., :1536]
  actff[token * KCAT + HID + j] = (_Float16)(g / (1.f + __expf(-g)) * xx);
}

// ---------------------------------------------------------------------------
// out[b][c][s] = acc[token][c] + residual x[b][c][s]  (channel-first restore)
// ---------------------------------------------------------------------------
__global__ __launch_bounds__(256) void final_kernel(
    const float* __restrict__ acc, const float* __restrict__ x,
    float* __restrict__ out) {
  const int b = blockIdx.x / HID;
  const int c = blockIdx.x % HID;
  const long bcs = ((long)b * HID + c) * SEQ;
  for (int s = threadIdx.x; s < SEQ; s += 256)
    out[bcs + s] = acc[((long)b * SEQ + s) * HID + c] + x[bcs + s];
}

// ---------------------------------------------------------------------------
// Host driver
// ---------------------------------------------------------------------------
extern "C" void kernel_launch(void* const* d_in, const int* in_sizes, int n_in,
                              void* d_out, int out_size, void* d_ws, size_t ws_size,
                              hipStream_t stream) {
  const float* x       = (const float*)d_in[0];
  const float* norm1_w = (const float*)d_in[1];
  const float* w_fused = (const float*)d_in[2];
  const float* b_fused = (const float*)d_in[3];
  const float* q_gamma = (const float*)d_in[4];
  const float* q_beta  = (const float*)d_in[5];
  const float* k_gamma = (const float*)d_in[6];
  const float* k_beta  = (const float*)d_in[7];
  const float* w_attn  = (const float*)d_in[8];
  const float* w_ff    = (const float*)d_in[9];
  const float* b_ff    = (const float*)d_in[10];
  float* out = (float*)d_out;

  // Workspace carve-up (256B aligned)
  size_t off = 0;
  auto carve = [&](size_t bytes) {
    size_t r = off;
    off = (off + bytes + 255) & ~(size_t)255;
    return r;
  };
  char* ws = (char*)d_ws;
  _Float16* wf16   = (_Float16*)(ws + carve((size_t)NFUSE * HID * 2));          // fused weights f16
  _Float16* wcat16 = (_Float16*)(ws + carve((size_t)HID * KCAT * 2));           // w_attn ++ w_ff
  _Float16* xn16   = (_Float16*)(ws + carve((size_t)TOK * HID * 2));            // normed activations
  float*    fusedf = (float*)   (ws + carve((size_t)TOK * NFUSE * 4));          // fused GEMM out
  _Float16* q16    = (_Float16*)(ws + carve((size_t)GB * HEADS * SEQ * HDIM * 2));
  _Float16* k16    = (_Float16*)(ws + carve((size_t)GB * HEADS * SEQ * HDIM * 2));
  _Float16* vT16   = (_Float16*)(ws + carve((size_t)GB * HEADS * HDIM * SEQ * 2));
  _Float16* probs  = (_Float16*)(ws + carve((size_t)GB * HEADS * SEQ * SEQ * 2));
  _Float16* actff  = (_Float16*)(ws + carve((size_t)TOK * KCAT * 2));           // att ++ silu*ff
  float*    accf   = (float*)   (ws + carve((size_t)TOK * HID * 4));            // pre-residual out

  const dim3 blk(256, 1, 1);

  // 1) weights -> f16 (w_attn and w_ff concatenated into one [768,2304] matrix)
  {
    long n = (long)NFUSE * HID;
    cvt_kernel<<<dim3((unsigned)((n + 255) / 256)), blk, 0, stream>>>(w_fused, wf16, NFUSE, HID, HID, 0);
    n = (long)HID * HID;
    cvt_kernel<<<dim3((unsigned)((n + 255) / 256)), blk, 0, stream>>>(w_attn, wcat16, HID, HID, KCAT, 0);
    n = (long)HID * (MLP / 2);
    cvt_kernel<<<dim3((unsigned)((n + 255) / 256)), blk, 0, stream>>>(w_ff, wcat16, HID, MLP / 2, KCAT, HID);
  }

  // 2) RMS norm + transpose to [token, C] f16
  rmsnorm_kernel<<<dim3(GB * HID), blk, 0, stream>>>(x, norm1_w, xn16);

  // 3) fused projection: [3456,768] x [5376,768]^T -> f32 [3456,5376]
  wmma_gemm<2, 4, false><<<dim3(NFUSE / 256, TOK / 64, 1), blk, 0, stream>>>(
      xn16, HID, 0, wf16, HID, 0, fusedf, NFUSE, 1, 0, 0, b_fused, TOK, NFUSE, HID);

  // 4) q/k LN + RoPE, v transpose; f16 outputs
  qkv_prep_kernel<<<dim3(TOK * HEADS / 8), blk, 0, stream>>>(
      fusedf, q_gamma, q_beta, k_gamma, k_beta, q16, k16, vT16);

  // 5) scores = (q/8) . k : batched [1728,64] x [1728,64]^T -> f16 [1728,1728]
  wmma_gemm<4, 2, true><<<dim3(14, 14, GB * HEADS), blk, 0, stream>>>(
      q16, HDIM, (long)SEQ * HDIM, k16, HDIM, (long)SEQ * HDIM,
      probs, SEQ, 1, (long)SEQ * SEQ, 0, nullptr, SEQ, SEQ, HDIM);

  // 6) row softmax in place
  softmax_kernel<<<dim3(GB * HEADS * SEQ), blk, 0, stream>>>(probs);

  // 7) att = P.V : batched [1728,1728] x [64,1728]^T -> f16 scattered into
  //    actff[token][head*64 + d]  (C offset = (z/12)*1728*2304 + (z%12)*64)
  wmma_gemm<8, 1, true><<<dim3(1, 7, GB * HEADS), blk, 0, stream>>>(
      probs, SEQ, (long)SEQ * SEQ, vT16, SEQ, (long)HDIM * SEQ,
      actff, KCAT, HEADS, (long)SEQ * KCAT, HDIM, nullptr, SEQ, HDIM, SEQ);

  // 8) SwiGLU fills actff[token][768:2304]
  {
    long n = (long)TOK * (MLP / 2);
    silu_kernel<<<dim3((unsigned)((n + 255) / 256)), blk, 0, stream>>>(fusedf, actff);
  }

  // 9) fused output projection: [3456,2304] x [768,2304]^T (+b_ff)
  //    == attn_out + ff_out in a single GEMM
  wmma_gemm<2, 4, false><<<dim3(HID / 256, TOK / 64, 1), blk, 0, stream>>>(
      actff, KCAT, 0, wcat16, KCAT, 0, accf, HID, 1, 0, 0, b_ff, TOK, HID, KCAT);

  // 10) residual add + channel-first transpose
  final_kernel<<<dim3(GB * HID), blk, 0, stream>>>(accf, x, out);
}
